// CompressedMoE_31550829757014
// MI455X (gfx1250) — compile-verified, matrix operand order checked by reading, not run on hardware
//
#include <hip/hip_runtime.h>

// ---------------------------------------------------------------------------
// CompressedMoE forward on MI455X (gfx1250):
//   out = X @ W0^T + b0      (router in the reference is dead code)
//   M=8192, N=2048, K=2048, fp32 in/out.  Compute-bound (AI ~ 477 FLOP/B);
//   whole working set (144 MB) fits in the 192 MB L2.
//
// Fast path (needs 80 MB workspace):
//   pass 1: stream-split fp32 -> bf16 hi/lo once (Xhi/Xlo/Whi/Wlo in d_ws)
//   pass 2: bf16x3 WMMA GEMM, tiles staged via GLOBAL_LOAD_ASYNC_TO_LDS_B128
//           (ASYNCcnt), double-buffered LDS, DMA overlapped with WMMA.
// Fallback: fused kernel converting in-loop (no workspace assumption).
// ---------------------------------------------------------------------------

typedef __attribute__((ext_vector_type(16))) __bf16 v16bf;
typedef __attribute__((ext_vector_type(8)))  float  v8f;

#define M_DIM 8192
#define N_DIM 2048
#define K_DIM 2048

#define BM 128          // block tile M
#define BN 128          // block tile N
#define BK 64           // K elements per LDS stage (2 WMMA k-steps)
#define ROWB (BK * 2)   // bytes per LDS tile row (64 bf16 = 128 B)

#define TILE_B   (BM * ROWB)   // 16 KB per tile
#define OFF_AHI  (0 * TILE_B)
#define OFF_ALO  (1 * TILE_B)
#define OFF_BHI  (2 * TILE_B)
#define OFF_BLO  (3 * TILE_B)
#define LDS_BUF  (4 * TILE_B)  // 64 KB per stage buffer

__device__ __forceinline__ void wmma_bf16(v16bf a, v16bf b, v8f& c) {
    c = __builtin_amdgcn_wmma_f32_16x16x32_bf16(
            /*neg_a=*/false, a, /*neg_b=*/false, b,
            /*c_mod=*/(short)0, c, /*reuse_a=*/false, /*reuse_b=*/false);
}

// WMMA 16-bit operand fragment from a swizzled LDS tile.
// lanes 0-15 : row = lane,    K 16B-groups {0,2} of the 32-wide k-step
// lanes 16-31: row = lane-16, K 16B-groups {1,3}
__device__ __forceinline__ v16bf frag_ld(const unsigned char* base, int row,
                                         int ks, int half) {
    const unsigned s  = ((unsigned)row >> 1) & 7u;
    const unsigned c0 = (unsigned)((ks << 2) + half);
    const unsigned c1 = c0 + 2u;
    union { uint4 q[2]; v16bf v; } U;
    U.q[0] = *(const uint4*)(base + (unsigned)row * ROWB + ((c0 ^ s) << 4));
    U.q[1] = *(const uint4*)(base + (unsigned)row * ROWB + ((c1 ^ s) << 4));
    return U.v;
}

// ===========================================================================
// Pass 1: fp32 -> bf16 hi/lo split (elementwise stream, float4 granularity)
// ===========================================================================
__global__ __launch_bounds__(256)
void split_bf16_kernel(const float* __restrict__ X, const float* __restrict__ W,
                       __bf16* __restrict__ Xhi, __bf16* __restrict__ Xlo,
                       __bf16* __restrict__ Whi, __bf16* __restrict__ Wlo)
{
    const size_t nX4 = (size_t)M_DIM * K_DIM / 4;   // 4,194,304
    const size_t gid = (size_t)blockIdx.x * 256 + threadIdx.x;

    const float* src;
    __bf16 *dh, *dl;
    size_t idx4;
    if (gid < nX4) { src = X; dh = Xhi; dl = Xlo; idx4 = gid; }
    else           { src = W; dh = Whi; dl = Wlo; idx4 = gid - nX4; }

    float4 v = ((const float4*)src)[idx4];
    union { __bf16 b[4]; uint2 u; } h, l;
    const float* f = (const float*)&v;
#pragma unroll
    for (int e = 0; e < 4; ++e) {
        float x = f[e];
        __bf16 hi = (__bf16)x;
        h.b[e] = hi;
        l.b[e] = (__bf16)(x - (float)hi);
    }
    ((uint2*)dh)[idx4] = h.u;
    ((uint2*)dl)[idx4] = l.u;
}

// ===========================================================================
// Pass 2: bf16x3 GEMM, async global->LDS staging, double-buffered
// ===========================================================================
__global__ __launch_bounds__(256)
void moe_gemm_async_bf16x3(const __bf16* __restrict__ Xhi, const __bf16* __restrict__ Xlo,
                           const __bf16* __restrict__ Whi, const __bf16* __restrict__ Wlo,
                           const float* __restrict__ bias, float* __restrict__ O)
{
    extern __shared__ __align__(16) unsigned char dynlds[];   // 2 x 64 KB

    const int t     = threadIdx.x;
    const int lane  = t & 31;
    const int wave  = t >> 5;
    const int waveM = wave & 1;
    const int waveN = wave >> 1;
    const int l16   = lane & 15;
    const int half  = (lane >> 4) & 1;

    const int bm = blockIdx.y * BM;
    const int bn = blockIdx.x * BN;

    v8f acc[4][2];
#pragma unroll
    for (int i = 0; i < 4; ++i)
#pragma unroll
        for (int j = 0; j < 2; ++j)
            acc[i][j] = (v8f)(0.0f);

    // ---- async DMA issue: 16 x global_load_async_to_lds_b128 per thread ----
    // chunk id = t + 256*j per tile: row = (t>>3)+32*j, 16B-chunk c = t&7.
    // swizzle s = ((row>>1)&7) is invariant in j (row step 32), so LDS dest
    // advances by +4096 B/j while global advances by +128 KB/j.
    const int row0    = t >> 3;                 // 0..31
    const int c0      = t & 7;
    const unsigned sw = (((unsigned)row0 >> 1) & 7u);
    const unsigned lchunk = (unsigned)row0 * ROWB + ((((unsigned)c0) ^ sw) << 4);

    auto issue_stage = [&](int kt, int buf) {
        const unsigned l0 = (unsigned)buf * LDS_BUF + lchunk;
        const size_t gA = (size_t)(bm + row0) * K_DIM + (size_t)kt * BK + (size_t)c0 * 8;
        const size_t gB = (size_t)(bn + row0) * K_DIM + (size_t)kt * BK + (size_t)c0 * 8;
        const unsigned long long pAhi = (unsigned long long)(Xhi + gA);
        const unsigned long long pAlo = (unsigned long long)(Xlo + gA);
        const unsigned long long pBhi = (unsigned long long)(Whi + gB);
        const unsigned long long pBlo = (unsigned long long)(Wlo + gB);
#pragma unroll
        for (int j = 0; j < 4; ++j) {
            const unsigned           ld = l0 + (unsigned)j * 4096u;
            const unsigned long long gs = (unsigned long long)j * (32ull * K_DIM * 2ull);
            asm volatile("global_load_async_to_lds_b128 %0, %1, off"
                         :: "v"(ld + OFF_AHI), "v"(pAhi + gs) : "memory");
            asm volatile("global_load_async_to_lds_b128 %0, %1, off"
                         :: "v"(ld + OFF_ALO), "v"(pAlo + gs) : "memory");
            asm volatile("global_load_async_to_lds_b128 %0, %1, off"
                         :: "v"(ld + OFF_BHI), "v"(pBhi + gs) : "memory");
            asm volatile("global_load_async_to_lds_b128 %0, %1, off"
                         :: "v"(ld + OFF_BLO), "v"(pBlo + gs) : "memory");
        }
    };

    const int NKT = K_DIM / BK;   // 32 stages
    issue_stage(0, 0);

    for (int kt = 0; kt < NKT; ++kt) {
        // my DMA done -> barrier -> everyone's stage buffer is readable
        asm volatile("s_wait_asynccnt 0x0" ::: "memory");
        __syncthreads();

        if (kt + 1 < NKT) issue_stage(kt + 1, (kt + 1) & 1);  // overlap with WMMA

        const unsigned char* buf = dynlds + (unsigned)(kt & 1) * LDS_BUF;
#pragma unroll
        for (int ks = 0; ks < 2; ++ks) {
            v16bf bhi[2], blo[2];
#pragma unroll
            for (int nt = 0; nt < 2; ++nt) {
                const int r = waveN * 32 + nt * 16 + l16;
                bhi[nt] = frag_ld(buf + OFF_BHI, r, ks, half);
                blo[nt] = frag_ld(buf + OFF_BLO, r, ks, half);
            }
#pragma unroll
            for (int mt = 0; mt < 4; ++mt) {
                const int r = waveM * 64 + mt * 16 + l16;
                v16bf ahi = frag_ld(buf + OFF_AHI, r, ks, half);
                v16bf alo = frag_ld(buf + OFF_ALO, r, ks, half);
#pragma unroll
                for (int nt = 0; nt < 2; ++nt) {
                    wmma_bf16(ahi, bhi[nt], acc[mt][nt]);   // hi*hi
                    wmma_bf16(ahi, blo[nt], acc[mt][nt]);   // hi*lo
                    wmma_bf16(alo, bhi[nt], acc[mt][nt]);   // lo*hi
                }
            }
        }
        __syncthreads();   // readers done before this buffer's next DMA lands
    }

    // ---- epilogue: + bias, store f32 ----
    const int n_base = bn + waveN * 32 + l16;
    const int m_base = bm + waveM * 64 + (half << 3);
    float bv[2];
#pragma unroll
    for (int nt = 0; nt < 2; ++nt) bv[nt] = bias[n_base + nt * 16];

#pragma unroll
    for (int mt = 0; mt < 4; ++mt)
#pragma unroll
        for (int nt = 0; nt < 2; ++nt)
#pragma unroll
            for (int v = 0; v < 8; ++v)
                O[(size_t)(m_base + mt * 16 + v) * N_DIM + (n_base + nt * 16)] =
                    acc[mt][nt][v] + bv[nt];
}

// ===========================================================================
// Fallback: fused in-loop split (no workspace requirement)
// ===========================================================================
__global__ __launch_bounds__(256)
void moe_gemm_fused_bf16x3(const float* __restrict__ X,
                           const float* __restrict__ W,
                           const float* __restrict__ bias,
                           float* __restrict__ O)
{
    __shared__ __align__(16) unsigned char lds[4 * BM * ROWB];
    unsigned char* sAhi = lds + OFF_AHI;
    unsigned char* sAlo = lds + OFF_ALO;
    unsigned char* sBhi = lds + OFF_BHI;
    unsigned char* sBlo = lds + OFF_BLO;

    const int t     = threadIdx.x;
    const int lane  = t & 31;
    const int wave  = t >> 5;
    const int waveM = wave & 1;
    const int waveN = wave >> 1;
    const int l16   = lane & 15;
    const int half  = (lane >> 4) & 1;

    const int bm = blockIdx.y * BM;
    const int bn = blockIdx.x * BN;

    float4 ra[8], rb[8];
    v8f acc[4][2];
#pragma unroll
    for (int i = 0; i < 4; ++i)
#pragma unroll
        for (int j = 0; j < 2; ++j)
            acc[i][j] = (v8f)(0.0f);

    auto g_load = [&](int kt) {
#pragma unroll
        for (int j = 0; j < 8; ++j) {
            const int chunk = t + 256 * j;
            const int row   = chunk >> 4;
            const int c4    = chunk & 15;
            ra[j] = ((const float4*)(X + (size_t)(bm + row) * K_DIM + (size_t)kt * BK))[c4];
            rb[j] = ((const float4*)(W + (size_t)(bn + row) * K_DIM + (size_t)kt * BK))[c4];
        }
    };

    auto lds_fill = [&]() {
#pragma unroll
        for (int j = 0; j < 8; ++j) {
            const int chunk = t + 256 * j;
            const int row   = chunk >> 4;
            const int c4    = chunk & 15;
            const unsigned off = (unsigned)row * ROWB +
                                 ((((unsigned)(c4 >> 1)) ^ (((unsigned)row >> 1) & 7u)) << 4) +
                                 (((unsigned)c4 & 1u) << 3);
            union { __bf16 b[4]; uint2 u; } ph, pl, qh, ql;
            const float* fa = (const float*)&ra[j];
            const float* fb = (const float*)&rb[j];
#pragma unroll
            for (int e = 0; e < 4; ++e) {
                float xa = fa[e];
                __bf16 ha = (__bf16)xa;
                ph.b[e] = ha;
                pl.b[e] = (__bf16)(xa - (float)ha);
                float xb = fb[e];
                __bf16 hb = (__bf16)xb;
                qh.b[e] = hb;
                ql.b[e] = (__bf16)(xb - (float)hb);
            }
            *(uint2*)(sAhi + off) = ph.u;
            *(uint2*)(sAlo + off) = pl.u;
            *(uint2*)(sBhi + off) = qh.u;
            *(uint2*)(sBlo + off) = ql.u;
        }
    };

    const int NKT = K_DIM / BK;
    g_load(0);

    for (int kt = 0; kt < NKT; ++kt) {
        __syncthreads();
        lds_fill();
        __syncthreads();
        if (kt + 1 < NKT) g_load(kt + 1);

#pragma unroll
        for (int ks = 0; ks < 2; ++ks) {
            v16bf bhi[2], blo[2];
#pragma unroll
            for (int nt = 0; nt < 2; ++nt) {
                const int r = waveN * 32 + nt * 16 + l16;
                bhi[nt] = frag_ld(sBhi, r, ks, half);
                blo[nt] = frag_ld(sBlo, r, ks, half);
            }
#pragma unroll
            for (int mt = 0; mt < 4; ++mt) {
                const int r = waveM * 64 + mt * 16 + l16;
                v16bf ahi = frag_ld(sAhi, r, ks, half);
                v16bf alo = frag_ld(sAlo, r, ks, half);
#pragma unroll
                for (int nt = 0; nt < 2; ++nt) {
                    wmma_bf16(ahi, bhi[nt], acc[mt][nt]);
                    wmma_bf16(ahi, blo[nt], acc[mt][nt]);
                    wmma_bf16(alo, bhi[nt], acc[mt][nt]);
                }
            }
        }
    }

    const int n_base = bn + waveN * 32 + l16;
    const int m_base = bm + waveM * 64 + (half << 3);
    float bv[2];
#pragma unroll
    for (int nt = 0; nt < 2; ++nt) bv[nt] = bias[n_base + nt * 16];

#pragma unroll
    for (int mt = 0; mt < 4; ++mt)
#pragma unroll
        for (int nt = 0; nt < 2; ++nt)
#pragma unroll
            for (int v = 0; v < 8; ++v)
                O[(size_t)(m_base + mt * 16 + v) * N_DIM + (n_base + nt * 16)] =
                    acc[mt][nt][v] + bv[nt];
}

// ===========================================================================
extern "C" void kernel_launch(void* const* d_in, const int* in_sizes, int n_in,
                              void* d_out, int out_size, void* d_ws, size_t ws_size,
                              hipStream_t stream) {
    (void)in_sizes; (void)n_in; (void)out_size;
    const float* X  = (const float*)d_in[0];   // (4,2048,2048)
    // d_in[1] = router_w : dead code in the reference, never read
    const float* W0 = (const float*)d_in[2];   // expert_w: expert 0 = first D*D slab
    const float* b0 = (const float*)d_in[3];   // expert_b: expert 0 = first D
    float* O = (float*)d_out;

    const size_t nX = (size_t)M_DIM * K_DIM;   // 16,777,216
    const size_t nW = (size_t)N_DIM * K_DIM;   //  4,194,304
    const size_t need = (nX + nW) * 2 /*hi+lo*/ * 2 /*bytes*/;   // 83,886,080

    dim3 grid(N_DIM / BN, M_DIM / BM);         // 16 x 64

    if (ws_size >= need) {
        __bf16* Xhi = (__bf16*)d_ws;
        __bf16* Xlo = Xhi + nX;
        __bf16* Whi = Xlo + nX;
        __bf16* Wlo = Whi + nW;

        const int nBlk = (int)((nX + nW) / 4 / 256);   // 20480
        split_bf16_kernel<<<nBlk, 256, 0, stream>>>(X, W0, Xhi, Xlo, Whi, Wlo);
        moe_gemm_async_bf16x3<<<grid, 256, 2 * LDS_BUF, stream>>>(
            Xhi, Xlo, Whi, Wlo, b0, O);
    } else {
        moe_gemm_fused_bf16x3<<<grid, 256, 0, stream>>>(X, W0, b0, O);
    }
}
